// Attention_77627238908749
// MI455X (gfx1250) — matrix operand order repaired and checked by reading, not compiled, not run on hardware
//
#include <hip/hip_runtime.h>
#include <hip/hip_bf16.h>

#define T_DIM 64
#define B_DIM 8
#define S_DIM 512
#define H_DIM 512

typedef __attribute__((ext_vector_type(16))) __bf16        v16bf;
typedef __attribute__((ext_vector_type(8)))  float         v8f;
typedef __attribute__((ext_vector_type(4)))  unsigned int  u32x4;

__device__ __forceinline__ unsigned short f32_to_bf16(float f) {
  unsigned int u = __float_as_uint(f);
  unsigned int r = u + 0x7FFFu + ((u >> 16) & 1u);   // round-to-nearest-even
  return (unsigned short)(r >> 16);
}

__device__ __forceinline__ float fast_tanh(float x) {
#if __has_builtin(__builtin_amdgcn_tanhf)
  return __builtin_amdgcn_tanhf(x);      // gfx1250 v_tanh_f32 (TRANS pipe)
#else
  return tanhf(x);
#endif
}

// gfx1250 async global->LDS copies, tracked by ASYNCcnt (no VGPR round-trip).
__device__ __forceinline__ void async_b128(unsigned lds_byte_off, const void* gaddr) {
  asm volatile("global_load_async_to_lds_b128 %0, %1, off"
               :: "v"(lds_byte_off), "v"(gaddr) : "memory");
}
__device__ __forceinline__ void async_b32(unsigned lds_byte_off, const void* gaddr) {
  asm volatile("global_load_async_to_lds_b32 %0, %1, off"
               :: "v"(lds_byte_off), "v"(gaddr) : "memory");
}
#define WAIT_ASYNC(n) asm volatile("s_wait_asynccnt " #n ::: "memory")

__device__ __forceinline__ unsigned lds_off32(const void* p) {
  // flat shared-aperture address: low 32 bits are the LDS offset (ISA 10.2)
  return (unsigned)(unsigned long long)p;
}

// ---------------------------------------------------------------- prep kernels

// dst[N][K] bf16 = src[K][N] f32 (transpose for WMMA B operand staging)
__global__ void cvt_transpose_bf16(const float* __restrict__ src,
                                   unsigned short* __restrict__ dst,
                                   int Kd, int Nd) {
  int i = blockIdx.x * blockDim.x + threadIdx.x;   // over Nd*Kd, output-major
  if (i >= Nd * Kd) return;
  int n = i / Kd;
  int k = i % Kd;
  dst[i] = f32_to_bf16(src[(long long)k * Nd + n]);
}

// dec [T,B,H] -> dec_bf [B*T, H] and concat_bf[:, H:2H]
__global__ void prep_dec_kernel(const float* __restrict__ dec_in,
                                unsigned short* __restrict__ dec_bf,
                                unsigned short* __restrict__ concat_bf) {
  int i = blockIdx.x * blockDim.x + threadIdx.x;   // over T*B*H
  if (i >= T_DIM * B_DIM * H_DIM) return;
  int h = i % H_DIM;
  int b = (i / H_DIM) % B_DIM;
  int t = i / (B_DIM * H_DIM);
  unsigned short vv = f32_to_bf16(dec_in[i]);
  int row = b * T_DIM + t;
  dec_bf[row * H_DIM + h] = vv;
  concat_bf[row * (2 * H_DIM) + H_DIM + h] = vv;
}

// enc' = tanh(enc[s,b,h] + cov[b,s]*wcov[h])
//   -> enc_bf  [B*S, H]  (A operand of a2 GEMM)
//   -> enc_bfT [B][H][S] (pre-transposed B operand of context GEMM)
__global__ void prep_enc_kernel(const float* __restrict__ enc_in,
                                const float* __restrict__ cov,
                                const float* __restrict__ wcov,
                                unsigned short* __restrict__ enc_bf,
                                unsigned short* __restrict__ enc_bfT) {
  int i = blockIdx.x * blockDim.x + threadIdx.x;   // over B*S*H as (b,s,h)
  if (i >= B_DIM * S_DIM * H_DIM) return;
  int h = i % H_DIM;
  int s = (i / H_DIM) % S_DIM;
  int b = i / (S_DIM * H_DIM);
  float e = enc_in[(s * B_DIM + b) * H_DIM + h];
  unsigned short vv = f32_to_bf16(fast_tanh(e + cov[b * S_DIM + s] * wcov[h]));
  enc_bf[i] = vv;
  enc_bfT[((long long)b * H_DIM + h) * S_DIM + s] = vv;
}

// ---------------------------------------------------------------- WMMA GEMM
// C[M,N] = A[M,K] * Bt[N,K]^T (+bias). bf16 in, f32 accumulate.
// Tile: BM=64, BN=128, BK=32; 8 waves (2x4), each wave 2x2 16x16 WMMA tiles.
// Double-buffered LDS fed by global_load_async_to_lds_b128; pipeline is
// unrolled by two so buffer indices are compile-time constants.

#define GBM 64
#define GBN 128
#define GBK 32
#define SAP 40   // padded tile row (elems): 80B rows keep b128 align + banks
#define SBP 40

__global__ __launch_bounds__(256)
void gemm_bf16_wmma(const unsigned short* __restrict__ A, long long sAz, int lda,
                    const unsigned short* __restrict__ Bt, long long sBz, int ldbt,
                    const float* __restrict__ bias,
                    float* __restrict__ Cf, long long sCfz, int ldcf,
                    unsigned short* __restrict__ Cb, long long sCbz, int ldcb,
                    int K) {
  __shared__ unsigned short sA[2][GBM * SAP];
  __shared__ unsigned short sB[2][GBN * SBP];

  const int tid = threadIdx.x;
  const int n0  = blockIdx.x * GBN;
  const int m0  = blockIdx.y * GBM;
  const int z   = blockIdx.z;
  A  += (long long)z * sAz;
  Bt += (long long)z * sBz;

  const int wave = tid >> 5;
  const int lane = tid & 31;
  const int wm   = wave >> 2;    // 0..1
  const int wn   = wave & 3;     // 0..3
  const int half = lane >> 4;
  const int lm   = lane & 15;

  v8f acc[2][2];
  for (int i = 0; i < 2; i++)
    for (int j = 0; j < 2; j++)
      for (int r = 0; r < 8; r++) acc[i][j][r] = 0.0f;

  // staging map: A tile 64x32 -> 1 b128/thread; B tile 128x32 -> 2 b128/thread
  const int am  = tid >> 2;            // 0..63
  const int ak  = (tid & 3) * 8;       // 0,8,16,24
  const int bn1 = 64 + am;             // 64..127

  const unsigned short* gA  = A  + (long long)(m0 + am) * lda + ak;
  const unsigned short* gB0 = Bt + (long long)(n0 + am) * ldbt + ak;
  const unsigned short* gB1 = Bt + (long long)(n0 + bn1) * ldbt + ak;

  unsigned dA[2], dB0[2], dB1[2];
#pragma unroll
  for (int p = 0; p < 2; p++) {
    dA[p]  = lds_off32(&sA[p][am * SAP + ak]);
    dB0[p] = lds_off32(&sB[p][am * SAP + ak]);
    dB1[p] = lds_off32(&sB[p][bn1 * SBP + ak]);
  }

  auto stage = [&](int p, int kt) {      // 3 async instructions per wave
    async_b128(dA[p],  gA  + kt);
    async_b128(dB0[p], gB0 + kt);
    async_b128(dB1[p], gB1 + kt);
  };
  auto compute = [&](int p) {
    union Frag { u32x4 q[2]; v16bf v; } fa[2], fb[2];
#pragma unroll
    for (int tm = 0; tm < 2; tm++) {
      int row = wm * 32 + tm * 16 + lm;
      fa[tm].q[0] = *reinterpret_cast<const u32x4*>(&sA[p][row * SAP + half * 8]);
      fa[tm].q[1] = *reinterpret_cast<const u32x4*>(&sA[p][row * SAP + half * 8 + 16]);
    }
#pragma unroll
    for (int tn = 0; tn < 2; tn++) {
      int col = wn * 32 + tn * 16 + lm;
      fb[tn].q[0] = *reinterpret_cast<const u32x4*>(&sB[p][col * SBP + half * 8]);
      fb[tn].q[1] = *reinterpret_cast<const u32x4*>(&sB[p][col * SBP + half * 8 + 16]);
    }
#pragma unroll
    for (int tm = 0; tm < 2; tm++)
#pragma unroll
      for (int tn = 0; tn < 2; tn++)
        acc[tm][tn] = __builtin_amdgcn_wmma_f32_16x16x32_bf16(
            false, fa[tm].v, false, fb[tn].v,
            (short)0, acc[tm][tn], false, false);
  };

  const int nt = K / GBK;                // 16 or 32 (even, >= 2)
  stage(0, 0);
  for (int t = 0; t < nt - 2; t += 2) {  // compile-time buffer indices
    stage(1, (t + 1) * GBK);
    WAIT_ASYNC(3);                       // tile t landed (in-order completion)
    __syncthreads();
    compute(0);
    __syncthreads();
    stage(0, (t + 2) * GBK);
    WAIT_ASYNC(3);
    __syncthreads();
    compute(1);
    __syncthreads();
  }
  stage(1, (nt - 1) * GBK);              // tail: tiles nt-2 (buf0), nt-1 (buf1)
  WAIT_ASYNC(3);
  __syncthreads();
  compute(0);
  WAIT_ASYNC(0);
  __syncthreads();
  compute(1);

  // epilogue (C layout: vgpr r -> row r + 8*half, col = lm)
#pragma unroll
  for (int tm = 0; tm < 2; tm++) {
#pragma unroll
    for (int tn = 0; tn < 2; tn++) {
      int gcol = n0 + wn * 32 + tn * 16 + lm;
      float bv = bias ? bias[gcol] : 0.0f;
#pragma unroll
      for (int r = 0; r < 8; r++) {
        int grow = m0 + wm * 32 + tm * 16 + half * 8 + r;
        float val = acc[tm][tn][r] + bv;
        if (Cf) Cf[(long long)z * sCfz + (long long)grow * ldcf + gcol] = val;
        if (Cb) Cb[(long long)z * sCbz + (long long)grow * ldcb + gcol] =
            f32_to_bf16(val);
      }
    }
  }
}

// ---------------------------------------------------------------- score kernel
// scores[b,t,s] = sum_h v[h] * tanh(a1[b,t,h] + a2[b,s,h])
// Block handles (b, 16 t, 64 s); h in chunks of 64, double-buffered through
// LDS via async loads: b128 for a1, per-lane scattered b32 for the [s][h] ->
// [h][s] transpose of a2 (async dest is an arbitrary per-lane LDS address).

#define SC_HK 64
#define SC_NC (H_DIM / SC_HK)   // 8 chunks (even)

__global__ __launch_bounds__(256)
void score_kernel(const float* __restrict__ a1, const float* __restrict__ a2,
                  const float* __restrict__ v, float* __restrict__ scores) {
  __shared__ float sh_a1[2][16 * SC_HK];   // [t][h] (wave-uniform reads)
  __shared__ float sh_a2[2][SC_HK * 65];   // [h][s], pad 65 vs bank conflicts
  __shared__ float sh_v[H_DIM];

  const int tid = threadIdx.x;
  const int s0  = blockIdx.x * 64;
  const int t0  = blockIdx.y * 16;
  const int b   = blockIdx.z;

  const int sl = tid & 63;            // local s (lane-contiguous)
  const int tb = (tid >> 6) * 4;      // local t base: 0,4,8,12

  sh_v[tid]       = v[tid];
  sh_v[tid + 256] = v[tid + 256];

  // a1: one b128 (4 floats) per thread per chunk
  const int a1t = tid >> 4;           // 0..15
  const int a1h = (tid & 15) * 4;     // 0..60
  const float* gA1 = a1 + (long long)(b * T_DIM + t0 + a1t) * H_DIM + a1h;
  const unsigned dA1[2] = { lds_off32(&sh_a1[0][a1t * SC_HK + a1h]),
                            lds_off32(&sh_a1[1][a1t * SC_HK + a1h]) };

  auto stage = [&](int p, int h0) {   // 17 async instructions per wave
    async_b128(dA1[p], gA1 + h0);
#pragma unroll
    for (int j = 0; j < 16; j++) {
      int lin = tid + j * 256;
      int ss = lin >> 6, hh = lin & 63;
      async_b32(lds_off32(&sh_a2[p][hh * 65 + ss]),
                a2 + (long long)(b * S_DIM + s0 + ss) * H_DIM + h0 + hh);
    }
  };

  float acc0 = 0.f, acc1 = 0.f, acc2 = 0.f, acc3 = 0.f;
  auto compute = [&](int p, int h0) {
#pragma unroll 4
    for (int hh = 0; hh < SC_HK; hh++) {
      float vh  = sh_v[h0 + hh];
      float a2v = sh_a2[p][hh * 65 + sl];
      acc0 += vh * fast_tanh(sh_a1[p][(tb + 0) * SC_HK + hh] + a2v);
      acc1 += vh * fast_tanh(sh_a1[p][(tb + 1) * SC_HK + hh] + a2v);
      acc2 += vh * fast_tanh(sh_a1[p][(tb + 2) * SC_HK + hh] + a2v);
      acc3 += vh * fast_tanh(sh_a1[p][(tb + 3) * SC_HK + hh] + a2v);
    }
  };

  stage(0, 0);
  for (int c = 0; c < SC_NC - 2; c += 2) {
    stage(1, (c + 1) * SC_HK);
    WAIT_ASYNC(17);                  // chunk c landed (in-order completion)
    __syncthreads();
    compute(0, c * SC_HK);
    __syncthreads();
    stage(0, (c + 2) * SC_HK);
    WAIT_ASYNC(17);
    __syncthreads();
    compute(1, (c + 1) * SC_HK);
    __syncthreads();
  }
  stage(1, (SC_NC - 1) * SC_HK);
  WAIT_ASYNC(17);
  __syncthreads();
  compute(0, (SC_NC - 2) * SC_HK);
  WAIT_ASYNC(0);
  __syncthreads();
  compute(1, (SC_NC - 1) * SC_HK);

  long long base = (long long)(b * T_DIM + t0 + tb) * S_DIM + s0 + sl;
  scores[base]             = acc0;
  scores[base + S_DIM]     = acc1;
  scores[base + 2 * S_DIM] = acc2;
  scores[base + 3 * S_DIM] = acc3;
}

// ---------------------------------------------------------------- softmax
// One wave32 per (b,t) row of 512. Writes transposed f32 align to d_out
// and a bf16 copy for the context GEMM.

__global__ __launch_bounds__(256)
void softmax_kernel(const float* __restrict__ scores,
                    float* __restrict__ align_out,        // [T][B][S]
                    unsigned short* __restrict__ align_bf) {
  const int wave = threadIdx.x >> 5;
  const int lane = threadIdx.x & 31;
  const int row  = blockIdx.x * 8 + wave;   // b*T + t
  const int b = row / T_DIM;
  const int t = row % T_DIM;

  float x[16];
  float m = -3.4e38f;
#pragma unroll
  for (int i = 0; i < 16; i++) {
    x[i] = scores[(long long)row * S_DIM + lane + i * 32];
    m = fmaxf(m, x[i]);
  }
#pragma unroll
  for (int off = 16; off > 0; off >>= 1)
    m = fmaxf(m, __shfl_xor(m, off, 32));
  float sum = 0.f;
#pragma unroll
  for (int i = 0; i < 16; i++) {
    x[i] = __expf(x[i] - m);
    sum += x[i];
  }
#pragma unroll
  for (int off = 16; off > 0; off >>= 1)
    sum += __shfl_xor(sum, off, 32);
  float inv = __builtin_amdgcn_rcpf(sum);
#pragma unroll
  for (int i = 0; i < 16; i++) {
    float val = x[i] * inv;
    int s = lane + i * 32;
    align_out[(long long)t * (B_DIM * S_DIM) + b * S_DIM + s] = val;
    align_bf[(long long)row * S_DIM + s] = f32_to_bf16(val);
  }
}

// ---------------------------------------------------------------- epilogue
__global__ void out_transpose_kernel(const float* __restrict__ attn_tmp,
                                     float* __restrict__ out) {
  int i = blockIdx.x * blockDim.x + threadIdx.x;   // (t,b,h) order
  if (i >= T_DIM * B_DIM * H_DIM) return;
  int h = i % H_DIM;
  int b = (i / H_DIM) % B_DIM;
  int t = i / (B_DIM * H_DIM);
  out[i] = attn_tmp[(long long)(b * T_DIM + t) * H_DIM + h];
}

// ---------------------------------------------------------------- launch

extern "C" void kernel_launch(void* const* d_in, const int* in_sizes, int n_in,
                              void* d_out, int out_size, void* d_ws, size_t ws_size,
                              hipStream_t stream) {
  (void)in_sizes; (void)n_in; (void)out_size; (void)ws_size;
  const float* dec_in = (const float*)d_in[0];
  const float* enc_in = (const float*)d_in[1];
  const float* cov    = (const float*)d_in[2];
  const float* Wq     = (const float*)d_in[3];
  const float* bq     = (const float*)d_in[4];
  const float* Wc     = (const float*)d_in[5];
  const float* v      = (const float*)d_in[6];
  const float* Wo     = (const float*)d_in[7];
  const float* bo     = (const float*)d_in[8];
  const float* wcov   = (const float*)d_in[9];
  float* out = (float*)d_out;

  char* ws = (char*)d_ws;
  size_t off = 0;
  auto alloc = [&](size_t bytes) -> void* {
    void* p = ws + off;
    off += (bytes + 255) & ~(size_t)255;
    return p;
  };

  unsigned short* dec_bf    = (unsigned short*)alloc((size_t)512 * 512 * 2);
  unsigned short* WqT_bf    = (unsigned short*)alloc((size_t)512 * 512 * 2);
  unsigned short* WcT_bf    = (unsigned short*)alloc((size_t)512 * 512 * 2);
  unsigned short* WoT_bf    = (unsigned short*)alloc((size_t)512 * 1024 * 2);
  unsigned short* enc_bf    = (unsigned short*)alloc((size_t)4096 * 512 * 2);
  unsigned short* enc_bfT   = (unsigned short*)alloc((size_t)4096 * 512 * 2);
  float*          a1        = (float*)alloc((size_t)512 * 512 * 4);
  float*          a2        = (float*)alloc((size_t)4096 * 512 * 4);
  float*          scores    = (float*)alloc((size_t)512 * 512 * 4);
  unsigned short* align_bf  = (unsigned short*)alloc((size_t)512 * 512 * 2);
  unsigned short* concat_bf = (unsigned short*)alloc((size_t)512 * 1024 * 2);
  float*          attn_tmp  = (float*)alloc((size_t)512 * 512 * 4);

  // weight transposes (B operands pre-laid-out as [N][K] bf16) + activations
  cvt_transpose_bf16<<<(512 * 512 + 255) / 256, 256, 0, stream>>>(Wq, WqT_bf, 512, 512);
  cvt_transpose_bf16<<<(512 * 512 + 255) / 256, 256, 0, stream>>>(Wc, WcT_bf, 512, 512);
  cvt_transpose_bf16<<<(512 * 1024 + 255) / 256, 256, 0, stream>>>(Wo, WoT_bf, 1024, 512);
  prep_dec_kernel<<<(T_DIM * B_DIM * H_DIM) / 256, 256, 0, stream>>>(dec_in, dec_bf, concat_bf);
  prep_enc_kernel<<<(B_DIM * S_DIM * H_DIM) / 256, 256, 0, stream>>>(enc_in, cov, wcov,
                                                                     enc_bf, enc_bfT);

  // a1 = dec @ Wq + bq      [512,512] x [512,512]
  gemm_bf16_wmma<<<dim3(512 / GBN, 512 / GBM, 1), 256, 0, stream>>>(
      dec_bf, 0, 512, WqT_bf, 0, 512, bq,
      a1, 0, 512, (unsigned short*)nullptr, 0, 0, 512);

  // a2 = enc' @ Wc          [4096,512] x [512,512]
  gemm_bf16_wmma<<<dim3(512 / GBN, 4096 / GBM, 1), 256, 0, stream>>>(
      enc_bf, 0, 512, WcT_bf, 0, 512, (const float*)nullptr,
      a2, 0, 512, (unsigned short*)nullptr, 0, 0, 512);

  // scores + softmax
  score_kernel<<<dim3(S_DIM / 64, T_DIM / 16, B_DIM), 256, 0, stream>>>(a1, a2, v, scores);
  softmax_kernel<<<512 / 8, 256, 0, stream>>>(scores, out + T_DIM * B_DIM * H_DIM, align_bf);

  // c = align @ enc' (batched over b), B pre-transposed = enc_bfT[b][H][S];
  // bf16 result written straight into concat[:, 0:512]
  gemm_bf16_wmma<<<dim3(512 / GBN, 64 / GBM, B_DIM), 256, 0, stream>>>(
      align_bf, (long long)T_DIM * S_DIM, S_DIM,
      enc_bfT, (long long)H_DIM * S_DIM, S_DIM, (const float*)nullptr,
      (float*)nullptr, 0, 0,
      concat_bf, (long long)T_DIM * 2 * H_DIM, 2 * H_DIM, S_DIM);

  // attn_h = concat @ Wo + bo   [512,1024] x [1024,512]
  gemm_bf16_wmma<<<dim3(512 / GBN, 512 / GBM, 1), 256, 0, stream>>>(
      concat_bf, 0, 2 * H_DIM, WoT_bf, 0, 1024, bo,
      attn_tmp, 0, H_DIM, (unsigned short*)nullptr, 0, 0, 2 * H_DIM);

  out_transpose_kernel<<<(T_DIM * B_DIM * H_DIM) / 256, 256, 0, stream>>>(attn_tmp, out);
}
